// Check_70789650973098
// MI455X (gfx1250) — matrix operand order repaired and checked by reading, not compiled, not run on hardware
//
#include <hip/hip_runtime.h>

// ---------------------------------------------------------------------------
// Bidirectional windowed LSTM (B=16,S=1024,E=256,H=256,FH=512,V=8000,WS=5)
//   1) gather embeddings -> bf16
//   2) Xproj = emb @ Wih^T + (bih+bhh) per direction (WMMA), row-padded so the
//      window indexing in the recurrence is branch-free (pad rows = bias)
//   3) 6-step LSTM recurrence, 32 samples / WG (2 m-tiles): each Whh B-frag
//      feeds 2 WMMAs, B loads batched 4-deep per clause, C starts from
//      inline-0 and Xproj is added at activation time, c in registers,
//      gates exchanged via LDS
//   4) head: leaky_relu -> fc1 (WMMA) -> leaky_relu -> fc2 dot -> sigmoid
// ---------------------------------------------------------------------------

#define Bv   16
#define Sv   1024
#define Ev   256
#define Hv   256
#define FHv  512
#define WSv  5
#define Nv   (Bv * Sv)            // 16384 samples
#define SP   (Sv + 2 * WSv)       // 1034 padded rows per batch
#define GC   1024                 // gate columns per direction (4*H)

typedef __attribute__((ext_vector_type(16))) __bf16 v16bf;
typedef __attribute__((ext_vector_type(8)))  float  v8f;
typedef __bf16 bf;

struct B32B { uint4 a, b; };      // 32-byte chunk

static __device__ __forceinline__ float bf2f(bf x) {
  unsigned short s = __builtin_bit_cast(unsigned short, x);
  unsigned u = ((unsigned)s) << 16;
  return __builtin_bit_cast(float, u);
}
static __device__ __forceinline__ bf f2bf(float f) {
  unsigned u = __builtin_bit_cast(unsigned, f);
  u += 0x7FFFu + ((u >> 16) & 1u);              // round-to-nearest-even
  unsigned short s = (unsigned short)(u >> 16);
  return __builtin_bit_cast(bf, s);
}
static __device__ __forceinline__ float sigm(float x) { return 1.0f / (1.0f + __expf(-x)); }
// tanh via one v_exp_f32: e = exp(-2|x|); (1-e)/(1+e) with sign restore
static __device__ __forceinline__ float fast_tanh(float x) {
  float e = __expf(-2.0f * fabsf(x));
  float r = (1.0f - e) / (1.0f + e);
  return copysignf(r, x);
}
static __device__ __forceinline__ float lrelu(float x) { return x > 0.0f ? x : 0.01f * x; }

// B fragment (K=32 x N=16, bf16), weights row-major [Nrows][K]:
// lane: n = lane%16; runs [kt*32+h*8, +8) and [kt*32+16+h*8, +8), h = lane>=16
static __device__ __forceinline__ v16bf loadBfrag(const bf* __restrict__ W, int K,
                                                  int row, int koff) {
  const uint4* p = (const uint4*)(W + (size_t)row * K + (koff - ((koff & 16) >> 1)));
  B32B r{p[0], p[2]};
  return __builtin_bit_cast(v16bf, r);
}
// A fragment (M=16 x K=32, bf16), X row-major, row offset in halves:
// lane: m = lane%16; runs [off1,off1+8) and [off1+16,off1+24)
static __device__ __forceinline__ v16bf loadAfrag(const bf* __restrict__ X,
                                                  size_t rowoff, int off1) {
  uint4 lo = *(const uint4*)(X + rowoff + off1);
  uint4 hi = *(const uint4*)(X + rowoff + off1 + 16);
  B32B r{lo, hi};
  return __builtin_bit_cast(v16bf, r);
}

#define WMMA_BF16(C, A, Bm)                                                     \
  (C) = __builtin_amdgcn_wmma_f32_16x16x32_bf16(false, (A), false, (Bm),        \
                                                (short)0, (C), false, false)

// ---------------------------------------------------------------------------
// small prep kernels
// ---------------------------------------------------------------------------
__global__ void k_f2bf(const float* __restrict__ s, bf* __restrict__ d, int n) {
  int i = blockIdx.x * 256 + threadIdx.x;
  if (i < n) d[i] = f2bf(s[i]);
}

__global__ void k_bias(const float* __restrict__ bihf, const float* __restrict__ bhhf,
                       const float* __restrict__ bihb, const float* __restrict__ bhhb,
                       float* __restrict__ bs) {
  int i = blockIdx.x * 256 + threadIdx.x;
  if (i < 2 * GC)
    bs[i] = (i < GC) ? (bihf[i] + bhhf[i]) : (bihb[i - GC] + bhhb[i - GC]);
}

__global__ void k_embed(const int* __restrict__ x, const float* __restrict__ tab,
                        bf* __restrict__ emb) {
  int n = blockIdx.x, e = threadIdx.x;
  int tok = x[n];
  float v = (tok == 0) ? 0.0f : tab[(size_t)tok * Ev + e];
  emb[(size_t)n * Ev + e] = f2bf(v);
}

// fill the 2*WS pad rows per batch of XprojPad with the bias sum
__global__ void k_padfill(const float* __restrict__ bs, bf* __restrict__ xp) {
  int blk = blockIdx.x;              // 16 batches * 10 pad rows
  int b = blk / (2 * WSv), pr = blk % (2 * WSv);
  int r = (pr < WSv) ? pr : (pr + Sv);
  size_t base = ((size_t)b * SP + r) * (2 * GC);
  for (int it = 0; it < 8; ++it) {
    int c = threadIdx.x + it * 256;
    xp[base + c] = f2bf(bs[c]);
  }
}

// ---------------------------------------------------------------------------
// Xproj GEMM: 16 samples x 2048 cols ([dir0|dir1]), K=256.
// kt-outer: A-frag reused by 16 n-tiles; B loads batched 4-deep; C starts
// at inline-0, bias folded in at the store epilogue.
// ---------------------------------------------------------------------------
__global__ __launch_bounds__(256) void k_xproj(const bf* __restrict__ emb,
                                               const bf* __restrict__ wihcat,
                                               const float* __restrict__ bs,
                                               bf* __restrict__ xp) {
  int n0 = blockIdx.x * 16;
  int b = n0 >> 10, t0 = n0 & (Sv - 1);
  int lane = threadIdx.x & 31, wave = threadIdx.x >> 5;
  int l16 = lane & 15, lh = lane >> 4;

  size_t arow = (size_t)(n0 + l16) * Ev;
  int ntb = wave * 16;                       // 16 n-tiles per wave

  v8f C[16] = {};
#pragma unroll
  for (int kt = 0; kt < 8; ++kt) {
    int koff = kt * 32 + lh * 16;
    v16bf a = loadAfrag(emb, arow, kt * 32 + lh * 8);
    v16bf barr[4];
#pragma unroll
    for (int g = 0; g < 4; ++g) {
#pragma unroll
      for (int i = 0; i < 4; ++i)
        barr[i] = loadBfrag(wihcat, Ev, (ntb + g * 4 + i) * 16 + l16, koff);
#pragma unroll
      for (int i = 0; i < 4; ++i)
        WMMA_BF16(C[g * 4 + i], a, barr[i]);
    }
  }

  size_t prow = ((size_t)b * SP + t0 + WSv) * (2 * GC);
#pragma unroll
  for (int nt = 0; nt < 16; ++nt) {
    int col = (ntb + nt) * 16 + l16;
    float bb = bs[col];
#pragma unroll
    for (int v = 0; v < 8; ++v) {
      int mv = v + lh * 8;
      xp[prow + (size_t)mv * (2 * GC) + col] = f2bf(C[nt][v] + bb);
    }
  }
}

// ---------------------------------------------------------------------------
// LSTM recurrence: grid = (512 tiles of 32 samples, 2 dirs).
// Per wave/step: 8 n-tiles x 8 k-tiles x 2 m-tiles = 128 WMMA; each Whh B-frag
// feeds 2 WMMAs. B loads batched 4-deep per clause. C accumulates from
// inline-0; Xproj added at activation time (prefetched ahead).
// ---------------------------------------------------------------------------
#define HSTR 264   // h LDS row stride (bf16), 32 rows
#define GSTR 1040  // gates LDS row stride (bf16), 16 rows

__global__ __launch_bounds__(256) void k_lstm(const bf* __restrict__ whh_all,
                                              const bf* __restrict__ xp,
                                              bf* __restrict__ hcat) {
  __shared__ bf h_s[32 * HSTR];
  __shared__ bf g_s[16 * GSTR];

  int dir = blockIdx.y;
  int n0 = blockIdx.x * 32;
  int b = n0 >> 10, t0 = n0 & (Sv - 1);
  int tid = threadIdx.x, lane = tid & 31, wave = tid >> 5;
  int l16 = lane & 15, lh = lane >> 4;
  const bf* whh = whh_all + (size_t)dir * GC * Hv;
  int ntb = wave * 8;

  for (int i = tid; i < 32 * HSTR; i += 256) h_s[i] = f2bf(0.0f);
  float creg[32];
#pragma unroll
  for (int i = 0; i < 32; ++i) creg[i] = 0.0f;

  // prefetch step-0 Xproj rows
  {
    int mv2 = tid >> 3, ln = tid & 7;
    int r0 = t0 + mv2 + ((dir == 0) ? 0 : (2 * WSv));
    const bf* base = xp + ((size_t)b * SP + r0) * (2 * GC) + (size_t)dir * GC;
    __builtin_prefetch(base + ln * 64, 0, 3);
    __builtin_prefetch(base + (ln + 8) * 64, 0, 3);
  }
  __syncthreads();

  bool use_tanh = (wave == 4 || wave == 5);   // cols 512..767 = g gate

  for (int k = 0; k < 6; ++k) {
    // padded row base for this step: row(m) = b*SP + rowk + m
    int rowk = t0 + ((dir == 0) ? k : (2 * WSv - k));

    // ---- prefetch next step's Xproj rows (global_prefetch_b8) ----
    if (k < 5) {
      int mv2 = tid >> 3, ln = tid & 7;
      int rn = t0 + mv2 + ((dir == 0) ? (k + 1) : (2 * WSv - k - 1));
      const bf* base = xp + ((size_t)b * SP + rn) * (2 * GC) + (size_t)dir * GC;
      __builtin_prefetch(base + ln * 64, 0, 3);
      __builtin_prefetch(base + (ln + 8) * 64, 0, 3);
    }

    // ---- MMA: C from inline-0; kt-outer, B batched 4-deep ----
    v8f C[8][2] = {};
#pragma unroll
    for (int kt = 0; kt < 8; ++kt) {
      int off1 = kt * 32 + lh * 8;
      int koff = kt * 32 + lh * 16;
      uint4 lo0 = *(const uint4*)(h_s + l16 * HSTR + off1);
      uint4 hi0 = *(const uint4*)(h_s + l16 * HSTR + off1 + 16);
      uint4 lo1 = *(const uint4*)(h_s + (16 + l16) * HSTR + off1);
      uint4 hi1 = *(const uint4*)(h_s + (16 + l16) * HSTR + off1 + 16);
      B32B r0{lo0, hi0}, r1{lo1, hi1};
      v16bf a0 = __builtin_bit_cast(v16bf, r0);
      v16bf a1 = __builtin_bit_cast(v16bf, r1);

      v16bf barr[4];
#pragma unroll
      for (int g = 0; g < 2; ++g) {
#pragma unroll
        for (int i = 0; i < 4; ++i)
          barr[i] = loadBfrag(whh, Hv, (ntb + g * 4 + i) * 16 + l16, koff);
#pragma unroll
        for (int i = 0; i < 4; ++i) {
          WMMA_BF16(C[g * 4 + i][0], a0, barr[i]);
          WMMA_BF16(C[g * 4 + i][1], a1, barr[i]);
        }
      }
    }

    // ---- two m-tile phases: (+Xproj) -> act -> LDS -> c/h update ----
#pragma unroll
    for (int mt = 0; mt < 2; ++mt) {
#pragma unroll
      for (int nt = 0; nt < 8; ++nt) {
        int col = (ntb + nt) * 16 + l16;
        const bf* px = xp + ((size_t)b * SP + rowk + lh * 8 + mt * 16) * (2 * GC)
                          + (size_t)dir * GC + col;
#pragma unroll
        for (int v = 0; v < 8; ++v) {
          int mv = v + lh * 8;
          float x = C[nt][mt][v] + bf2f(px[(size_t)v * (2 * GC)]);
          g_s[mv * GSTR + col] = f2bf(use_tanh ? fast_tanh(x) : sigm(x));
        }
      }
      __syncthreads();
#pragma unroll
      for (int it = 0; it < 16; ++it) {
        int idx = tid + it * 256;
        int m = idx >> 8, j = idx & 255;
        float gi = bf2f(g_s[m * GSTR + j]);
        float gf = bf2f(g_s[m * GSTR + 256 + j]);
        float gg = bf2f(g_s[m * GSTR + 512 + j]);
        float go = bf2f(g_s[m * GSTR + 768 + j]);
        float c = gf * creg[mt * 16 + it] + gi * gg;
        creg[mt * 16 + it] = c;
        h_s[(mt * 16 + m) * HSTR + j] = f2bf(go * fast_tanh(c));
      }
      __syncthreads();
    }
  }

  // leaky_relu(h) -> Hcat[n][dir*256 + j]
  for (int it = 0; it < 32; ++it) {
    int idx = tid + it * 256;
    int m = idx >> 8, j = idx & 255;
    float h = bf2f(h_s[m * HSTR + j]);
    hcat[(size_t)(n0 + m) * (2 * Hv) + (size_t)dir * Hv + j] = f2bf(lrelu(h));
  }
}

// ---------------------------------------------------------------------------
// head: fc1 (512->512, WMMA, K=512) + leaky_relu + fc2 dot + sigmoid
// ---------------------------------------------------------------------------
#define H1STR 528
__global__ __launch_bounds__(256) void k_head(const bf* __restrict__ hcat,
                                              const bf* __restrict__ fc1w,
                                              const float* __restrict__ fc1b,
                                              const float* __restrict__ fc2w,
                                              const float* __restrict__ fc2b,
                                              float* __restrict__ out) {
  __shared__ float h1[16 * H1STR];
  __shared__ float part[256];
  int n0 = blockIdx.x * 16;
  int tid = threadIdx.x, lane = tid & 31, wave = tid >> 5;
  int l16 = lane & 15, lh = lane >> 4;
  int ntb = wave * 4;

  v8f C[4] = {};
  size_t arow = (size_t)(n0 + l16) * (2 * Hv);
#pragma unroll
  for (int kt = 0; kt < 16; ++kt) {
    int koff = kt * 32 + lh * 16;
    v16bf a = loadAfrag(hcat, arow, kt * 32 + lh * 8);
    v16bf barr[4];
#pragma unroll
    for (int i = 0; i < 4; ++i)
      barr[i] = loadBfrag(fc1w, 2 * Hv, (ntb + i) * 16 + l16, koff);
#pragma unroll
    for (int i = 0; i < 4; ++i)
      WMMA_BF16(C[i], a, barr[i]);
  }
#pragma unroll
  for (int nt = 0; nt < 4; ++nt) {
    int col = (ntb + nt) * 16 + l16;
    float bb = fc1b[col];
#pragma unroll
    for (int v = 0; v < 8; ++v) {
      int mv = v + lh * 8;
      h1[mv * H1STR + col] = lrelu(C[nt][v] + bb);
    }
  }
  __syncthreads();

  int m = tid >> 4, seg = tid & 15;
  float s = 0.0f;
#pragma unroll
  for (int c = 0; c < 32; ++c) {
    int jj = seg * 32 + c;
    s += h1[m * H1STR + jj] * fc2w[jj];
  }
  part[tid] = s;
  __syncthreads();
  if (tid < 16) {
    float tot = fc2b[0];
#pragma unroll
    for (int sg = 0; sg < 16; ++sg) tot += part[tid * 16 + sg];
    out[n0 + tid] = sigm(tot);
  }
}

// ---------------------------------------------------------------------------
extern "C" void kernel_launch(void* const* d_in, const int* in_sizes, int n_in,
                              void* d_out, int out_size, void* d_ws, size_t ws_size,
                              hipStream_t stream) {
  const int*   x      = (const int*)  d_in[0];
  const float* embed  = (const float*)d_in[1];
  const float* Wih_f  = (const float*)d_in[2];
  const float* Whh_f  = (const float*)d_in[3];
  const float* bih_f  = (const float*)d_in[4];
  const float* bhh_f  = (const float*)d_in[5];
  const float* Wih_b  = (const float*)d_in[6];
  const float* Whh_b  = (const float*)d_in[7];
  const float* bih_b  = (const float*)d_in[8];
  const float* bhh_b  = (const float*)d_in[9];
  const float* fc1_w  = (const float*)d_in[10];
  const float* fc1_b  = (const float*)d_in[11];
  const float* fc2_w  = (const float*)d_in[12];
  const float* fc2_b  = (const float*)d_in[13];
  float* out = (float*)d_out;

  // workspace carve-up (~95.6 MB)
  char* w = (char*)d_ws;
  bf*    wihcat = (bf*)w;                 w += (size_t)2 * GC * Ev * 2;
  bf*    whh    = (bf*)w;                 w += (size_t)2 * GC * Hv * 2;
  bf*    fc1bf  = (bf*)w;                 w += (size_t)FHv * 2 * Hv * 2;
  float* bs     = (float*)w;              w += (size_t)2 * GC * 4;
  bf*    emb    = (bf*)w;                 w += (size_t)Nv * Ev * 2;
  bf*    xproj  = (bf*)w;                 w += (size_t)Bv * SP * 2 * GC * 2;
  bf*    hcat   = (bf*)w;                 w += (size_t)Nv * 2 * Hv * 2;
  (void)ws_size; (void)n_in; (void)in_sizes; (void)out_size;

  const int CW = GC * Ev;  // 262144 elements per weight matrix
  k_f2bf<<<(CW + 255) / 256, 256, 0, stream>>>(Wih_f, wihcat, CW);
  k_f2bf<<<(CW + 255) / 256, 256, 0, stream>>>(Wih_b, wihcat + CW, CW);
  k_f2bf<<<(CW + 255) / 256, 256, 0, stream>>>(Whh_f, whh, CW);
  k_f2bf<<<(CW + 255) / 256, 256, 0, stream>>>(Whh_b, whh + CW, CW);
  k_f2bf<<<(CW + 255) / 256, 256, 0, stream>>>(fc1_w, fc1bf, CW);
  k_bias<<<8, 256, 0, stream>>>(bih_f, bhh_f, bih_b, bhh_b, bs);
  k_embed<<<Nv, 256, 0, stream>>>(x, embed, emb);
  k_padfill<<<Bv * 2 * WSv, 256, 0, stream>>>(bs, xproj);
  k_xproj<<<Nv / 16, 256, 0, stream>>>(emb, wihcat, bs, xproj);
  k_lstm<<<dim3(Nv / 32, 2), 256, 0, stream>>>(whh, xproj, hcat);
  k_head<<<Nv / 16, 256, 0, stream>>>(hcat, fc1bf, fc1_b, fc2_w, fc2_b, out);
}
